// KimiDeltaAttention_67001489818058
// MI455X (gfx1250) — matrix operand order repaired and checked by reading, not compiled, not run on hardware
//
#include <hip/hip_runtime.h>
#include <hip/hip_bf16.h>

// ---------------------------------------------------------------------------
// KimiDeltaAttention for MI455X (gfx1250, wave32, WMMA).
// T=2048, HID=2048, H=16, D=64, Kconv=4, PROJ=1024.
// Heavy GEMMs run as bf16 WMMA (v_wmma_f32_16x16x32_bf16), with LDS tiles
// staged through the CDNA5 async path (global_load_async_to_lds_b128,
// ASYNCcnt). Sequential delta-rule scan is a per-head register-state kernel.
// ---------------------------------------------------------------------------

#define T_LEN 2048
#define HID   2048
#define NH    16
#define HD    64
#define PROJ  1024
#define KCONV 4

typedef __bf16 bf16x8  __attribute__((ext_vector_type(8)));
typedef __bf16 bf16x16 __attribute__((ext_vector_type(16)));
typedef float  f32x8   __attribute__((ext_vector_type(8)));

// ----------------------------- fp32 -> bf16 --------------------------------
__global__ void kda_cvt_bf16(const float* __restrict__ src,
                             __bf16* __restrict__ dst, int n) {
    int i = blockIdx.x * blockDim.x + threadIdx.x;
    if (i < n) dst[i] = (__bf16)src[i];
}

// ----------------------- NT bf16 WMMA GEMM ---------------------------------
// C[m,n] = sum_k A[m,k] * B[n,k]   (A: MxK row-major, B: NxK row-major)
// Block = 256 threads = 8 waves (2x4). Block tile: 64(M) x 128(N), K slab 64.
// Each wave owns a 32x32 register tile (4 accumulators, 4 WMMA per k-step).
#define KC        64
#define LDS_STR   72   // padded row stride (elements) to dodge bank conflicts

__global__ void kda_gemm_nt_wmma(const __bf16* __restrict__ A,
                                 const __bf16* __restrict__ B,
                                 float* __restrict__ C,
                                 int M, int N, int K) {
    __shared__ __attribute__((aligned(16))) __bf16 As[64  * LDS_STR];
    __shared__ __attribute__((aligned(16))) __bf16 Bs[128 * LDS_STR];

    const int tid  = threadIdx.x;        // 0..255
    const int wave = tid >> 5;           // 0..7
    const int lane = tid & 31;
    const int hi   = lane >> 4;          // lane half (ISA fragment layout)
    const int l15  = lane & 15;

    const int m0 = blockIdx.x * 64;
    const int n0 = blockIdx.y * 128;
    const int mw = (wave >> 2) * 32;     // wave M offset within block tile
    const int nw = (wave &  3) * 32;     // wave N offset within block tile

    f32x8 acc[2][2] = {};

    auto loadA = [&](int row, int kk) -> bf16x16 {
        bf16x8 lo = *(const bf16x8*)&As[row * LDS_STR + kk + hi * 8];
        bf16x8 hv = *(const bf16x8*)&As[row * LDS_STR + kk + 16 + hi * 8];
        bf16x16 f;
        #pragma unroll
        for (int j = 0; j < 8; ++j) { f[j] = lo[j]; f[j + 8] = hv[j]; }
        return f;
    };
    auto loadB = [&](int col, int kk) -> bf16x16 {
        bf16x8 lo = *(const bf16x8*)&Bs[col * LDS_STR + kk + hi * 8];
        bf16x8 hv = *(const bf16x8*)&Bs[col * LDS_STR + kk + 16 + hi * 8];
        bf16x16 f;
        #pragma unroll
        for (int j = 0; j < 8; ++j) { f[j] = lo[j]; f[j + 8] = hv[j]; }
        return f;
    };

    for (int k0 = 0; k0 < K; k0 += KC) {
        // ---- async stage A slab: 64 rows x 64 cols (512 chunks, 2/thread)
        #pragma unroll
        for (int i = 0; i < 2; ++i) {
            int c  = i * 256 + tid;
            int r  = c >> 3;
            int kc = c & 7;
            unsigned lds = (unsigned)(uintptr_t)&As[r * LDS_STR + kc * 8];
            unsigned long long g =
                (unsigned long long)(uintptr_t)(A + (size_t)(m0 + r) * K + k0 + kc * 8);
            asm volatile("global_load_async_to_lds_b128 %0, %1, off"
                         :: "v"(lds), "v"(g) : "memory");
        }
        // ---- async stage B slab: 128 rows x 64 cols (1024 chunks, 4/thread)
        // branchless OOB handling: clamp row (clamped rows feed only tiles
        // whose store is guarded off below)
        #pragma unroll
        for (int i = 0; i < 4; ++i) {
            int c  = i * 256 + tid;
            int r  = c >> 3;
            int kc = c & 7;
            int rg = n0 + r; if (rg > N - 1) rg = N - 1;
            unsigned lds = (unsigned)(uintptr_t)&Bs[r * LDS_STR + kc * 8];
            unsigned long long g =
                (unsigned long long)(uintptr_t)(B + (size_t)rg * K + k0 + kc * 8);
            asm volatile("global_load_async_to_lds_b128 %0, %1, off"
                         :: "v"(lds), "v"(g) : "memory");
        }
        asm volatile("s_wait_asynccnt 0x0" ::: "memory");
        __syncthreads();

        // speculative prefetch of next K slab into GL2
        if (k0 + KC < K) {
            int rp = n0 + (tid >> 1); if (rp > N - 1) rp = N - 1;
            __builtin_prefetch(B + (size_t)rp * K + k0 + KC + (tid & 1) * 32, 0, 1);
            if (tid < 128)
                __builtin_prefetch(A + (size_t)(m0 + (tid >> 1)) * K + k0 + KC + (tid & 1) * 32, 0, 1);
        }

        // ---- 2 k-steps x 4 WMMAs ----
        #pragma unroll
        for (int kk = 0; kk < KC; kk += 32) {
            bf16x16 af0 = loadA(mw + l15,      kk);
            bf16x16 af1 = loadA(mw + 16 + l15, kk);
            bf16x16 bf0 = loadB(nw + l15,      kk);
            bf16x16 bf1 = loadB(nw + 16 + l15, kk);
            acc[0][0] = __builtin_amdgcn_wmma_f32_16x16x32_bf16(
                false, af0, false, bf0, (short)0, acc[0][0], false, false);
            acc[0][1] = __builtin_amdgcn_wmma_f32_16x16x32_bf16(
                false, af0, false, bf1, (short)0, acc[0][1], false, false);
            acc[1][0] = __builtin_amdgcn_wmma_f32_16x16x32_bf16(
                false, af1, false, bf0, (short)0, acc[1][0], false, false);
            acc[1][1] = __builtin_amdgcn_wmma_f32_16x16x32_bf16(
                false, af1, false, bf1, (short)0, acc[1][1], false, false);
        }
        __syncthreads();
    }

    // ---- store: VGPR r, lane l -> (M = r + 8*hi, N = l&15) per 16x16 tile
    #pragma unroll
    for (int i = 0; i < 2; ++i) {
        #pragma unroll
        for (int j = 0; j < 2; ++j) {
            int nbase = n0 + nw + j * 16;
            if (nbase < N) {
                #pragma unroll
                for (int r = 0; r < 8; ++r) {
                    int m = m0 + mw + i * 16 + r + 8 * hi;
                    C[(size_t)m * N + nbase + l15] = acc[i][j][r];
                }
            }
        }
    }
}

// ------------------ causal depthwise conv + SiLU (+ L2 norm) ---------------
// block = (t*NH + h), 64 threads = channels of one head.
__global__ void kda_dwconv_silu(const float* __restrict__ pre,
                                const float* __restrict__ w,
                                float* __restrict__ out, int do_norm) {
    int b = blockIdx.x;
    int t = b >> 4;
    int h = b & 15;
    int d = threadIdx.x;
    int c = h * HD + d;

    float acc = 0.f;
    #pragma unroll
    for (int i = 0; i < KCONV; ++i) {
        int tt = t + i - (KCONV - 1);
        if (tt >= 0) acc += pre[(size_t)tt * PROJ + c] * w[c * KCONV + i];
    }
    float s = acc / (1.f + __expf(-acc));   // SiLU

    float val = s;
    if (do_norm) {
        __shared__ float red[HD];
        red[d] = s * s;
        __syncthreads();
        for (int o = 32; o >= 1; o >>= 1) {
            if (d < o) red[d] += red[d + o];
            __syncthreads();
        }
        float nrm = sqrtf(red[0]);
        val = s / fmaxf(nrm, 1e-6f);
    }
    out[(size_t)t * PROJ + c] = val;
}

// ----------------- per-channel log-decay gate finalize ---------------------
__global__ void kda_gate_finalize(const float* __restrict__ raw,
                                  const float* __restrict__ A_log,
                                  const float* __restrict__ dt_bias,
                                  float* __restrict__ g) {
    int i = blockIdx.x * blockDim.x + threadIdx.x;
    if (i >= T_LEN * PROJ) return;
    int c = i & (PROJ - 1);
    int h = c >> 6;
    float x = raw[i] + dt_bias[c];
    float sp = (x > 20.f) ? x : log1pf(__expf(x));   // softplus
    g[i] = -__expf(A_log[h]) * sp;
}

__global__ void kda_sigmoid(const float* __restrict__ in,
                            float* __restrict__ out, int n) {
    int i = blockIdx.x * blockDim.x + threadIdx.x;
    if (i < n) out[i] = 1.f / (1.f + __expf(-in[i]));
}

// ------------------------- delta-rule scan ---------------------------------
// one block per head, 64 lanes; lane v owns state column S[:, v] in registers.
__global__ void kda_scan(const float* __restrict__ q,
                         const float* __restrict__ k,
                         const float* __restrict__ v,
                         const float* __restrict__ g,
                         const float* __restrict__ beta,
                         float* __restrict__ o) {
    const int head = blockIdx.x;
    const int tid  = threadIdx.x;       // v index
    const float scale = 0.125f;         // D^-0.5

    float S[HD];
    #pragma unroll
    for (int i = 0; i < HD; ++i) S[i] = 0.f;

    __shared__ float ks[HD], qs[HD], gs[HD];

    for (int t = 0; t < T_LEN; ++t) {
        size_t base = (size_t)t * PROJ + head * HD;
        ks[tid] = k[base + tid];
        qs[tid] = q[base + tid];
        gs[tid] = g[base + tid];
        __syncthreads();

        float bt  = beta[(size_t)t * NH + head];
        float err = v[base + tid];
        #pragma unroll
        for (int kk = 0; kk < HD; ++kk) {
            S[kk] *= __expf(gs[kk]);        // diagonal decay on key dim
            err   -= ks[kk] * S[kk];        // v_t - k_t^T S
        }
        float out = 0.f;
        #pragma unroll
        for (int kk = 0; kk < HD; ++kk) {
            S[kk] += bt * ks[kk] * err;     // rank-1 delta update
            out   += qs[kk] * S[kk];        // q_t^T S
        }
        o[base + tid] = out * scale;
        __syncthreads();
    }
}

// ----------------------- gated RMSNorm (in place) --------------------------
__global__ void kda_rmsnorm_gate(float* __restrict__ o,
                                 const float* __restrict__ og,
                                 const float* __restrict__ nw) {
    int b = blockIdx.x;
    int t = b >> 4;
    int h = b & 15;
    int d = threadIdx.x;
    size_t idx = (size_t)t * PROJ + h * HD + d;

    float x = o[idx];
    __shared__ float red[HD];
    red[d] = x * x;
    __syncthreads();
    for (int off = 32; off >= 1; off >>= 1) {
        if (d < off) red[d] += red[d + off];
        __syncthreads();
    }
    float var = red[0] * (1.f / HD);
    float y = x * rsqrtf(var + 1e-6f) * nw[d];
    float ogv = og[idx];
    y *= 1.f / (1.f + __expf(-ogv));
    o[idx] = y;
}

// ---------------------------------------------------------------------------
extern "C" void kernel_launch(void* const* d_in, const int* in_sizes, int n_in,
                              void* d_out, int out_size, void* d_ws, size_t ws_size,
                              hipStream_t stream) {
    const float* h_in    = (const float*)d_in[0];
    const float* Wq      = (const float*)d_in[1];
    const float* Wk      = (const float*)d_in[2];
    const float* Wv      = (const float*)d_in[3];
    const float* conv_q  = (const float*)d_in[4];
    const float* conv_k  = (const float*)d_in[5];
    const float* conv_v  = (const float*)d_in[6];
    const float* A_log   = (const float*)d_in[7];
    const float* dt_bias = (const float*)d_in[8];
    const float* Wfa     = (const float*)d_in[9];
    const float* Wfb     = (const float*)d_in[10];
    const float* Wb      = (const float*)d_in[11];
    const float* Wga     = (const float*)d_in[12];
    const float* Wgb     = (const float*)d_in[13];
    const float* norm_w  = (const float*)d_in[14];
    const float* Wo      = (const float*)d_in[15];
    float* out = (float*)d_out;

    // bump allocator over workspace
    size_t off = 0;
    auto alloc = [&](size_t bytes) -> void* {
        void* p = (char*)d_ws + off;
        off += (bytes + 255) & ~(size_t)255;
        return p;
    };

    __bf16* h_bf   = (__bf16*)alloc((size_t)T_LEN * HID * 2);
    __bf16* wq_bf  = (__bf16*)alloc((size_t)PROJ * HID * 2);
    __bf16* wk_bf  = (__bf16*)alloc((size_t)PROJ * HID * 2);
    __bf16* wv_bf  = (__bf16*)alloc((size_t)PROJ * HID * 2);
    __bf16* wo_bf  = (__bf16*)alloc((size_t)HID * PROJ * 2);
    __bf16* wfa_bf = (__bf16*)alloc((size_t)HD * HID * 2);
    __bf16* wfb_bf = (__bf16*)alloc((size_t)PROJ * HD * 2);
    __bf16* wga_bf = (__bf16*)alloc((size_t)HD * HID * 2);
    __bf16* wgb_bf = (__bf16*)alloc((size_t)PROJ * HD * 2);
    __bf16* wb_bf  = (__bf16*)alloc((size_t)NH * HID * 2);

    float* qp = (float*)alloc((size_t)T_LEN * PROJ * 4);
    float* kp = (float*)alloc((size_t)T_LEN * PROJ * 4);
    float* vp = (float*)alloc((size_t)T_LEN * PROJ * 4);
    float* qn = (float*)alloc((size_t)T_LEN * PROJ * 4);
    float* kn = (float*)alloc((size_t)T_LEN * PROJ * 4);
    float* vn = (float*)alloc((size_t)T_LEN * PROJ * 4);

    float*  fa    = (float*)alloc((size_t)T_LEN * HD * 4);
    __bf16* fa_bf = (__bf16*)alloc((size_t)T_LEN * HD * 2);
    float*  ga    = (float*)alloc((size_t)T_LEN * HD * 4);
    __bf16* ga_bf = (__bf16*)alloc((size_t)T_LEN * HD * 2);
    float*  rawg  = (float*)alloc((size_t)T_LEN * PROJ * 4);
    float*  gbuf  = (float*)alloc((size_t)T_LEN * PROJ * 4);
    float*  ogbuf = (float*)alloc((size_t)T_LEN * PROJ * 4);
    float*  braw  = (float*)alloc((size_t)T_LEN * NH * 4);
    float*  bsig  = (float*)alloc((size_t)T_LEN * NH * 4);
    float*  obuf  = (float*)alloc((size_t)T_LEN * PROJ * 4);
    __bf16* o_bf  = (__bf16*)alloc((size_t)T_LEN * PROJ * 2);

    auto cvt = [&](const float* s, __bf16* d, int n) {
        kda_cvt_bf16<<<(n + 255) / 256, 256, 0, stream>>>(s, d, n);
    };
    auto gemm = [&](const __bf16* A, const __bf16* B, float* C, int M, int N, int K) {
        dim3 grid(M / 64, (N + 127) / 128);
        kda_gemm_nt_wmma<<<grid, 256, 0, stream>>>(A, B, C, M, N, K);
    };

    // 1) bf16 conversions
    cvt(h_in, h_bf, T_LEN * HID);
    cvt(Wq,  wq_bf,  PROJ * HID);
    cvt(Wk,  wk_bf,  PROJ * HID);
    cvt(Wv,  wv_bf,  PROJ * HID);
    cvt(Wo,  wo_bf,  HID * PROJ);
    cvt(Wfa, wfa_bf, HD * HID);
    cvt(Wfb, wfb_bf, PROJ * HD);
    cvt(Wga, wga_bf, HD * HID);
    cvt(Wgb, wgb_bf, PROJ * HD);
    cvt(Wb,  wb_bf,  NH * HID);

    // 2) q/k/v projections (WMMA)
    gemm(h_bf, wq_bf, qp, T_LEN, PROJ, HID);
    gemm(h_bf, wk_bf, kp, T_LEN, PROJ, HID);
    gemm(h_bf, wv_bf, vp, T_LEN, PROJ, HID);

    // 3) depthwise conv + SiLU (+ L2 norm for q,k)
    kda_dwconv_silu<<<T_LEN * NH, HD, 0, stream>>>(qp, conv_q, qn, 1);
    kda_dwconv_silu<<<T_LEN * NH, HD, 0, stream>>>(kp, conv_k, kn, 1);
    kda_dwconv_silu<<<T_LEN * NH, HD, 0, stream>>>(vp, conv_v, vn, 0);

    // 4) low-rank decay gate: g = -exp(A_log) * softplus(h@Wfa^T@Wfb^T + dt_bias)
    gemm(h_bf, wfa_bf, fa, T_LEN, HD, HID);
    cvt(fa, fa_bf, T_LEN * HD);
    gemm(fa_bf, wfb_bf, rawg, T_LEN, PROJ, HD);
    kda_gate_finalize<<<(T_LEN * PROJ + 255) / 256, 256, 0, stream>>>(rawg, A_log, dt_bias, gbuf);

    // 5) output gate: og = h@Wga^T@Wgb^T (sigmoid fused into rmsnorm kernel)
    gemm(h_bf, wga_bf, ga, T_LEN, HD, HID);
    cvt(ga, ga_bf, T_LEN * HD);
    gemm(ga_bf, wgb_bf, ogbuf, T_LEN, PROJ, HD);

    // 6) beta = sigmoid(h @ Wb^T)
    gemm(h_bf, wb_bf, braw, T_LEN, NH, HID);
    kda_sigmoid<<<(T_LEN * NH + 255) / 256, 256, 0, stream>>>(braw, bsig, T_LEN * NH);

    // 7) sequential delta-rule scan
    kda_scan<<<NH, HD, 0, stream>>>(qn, kn, vn, gbuf, bsig, obuf);

    // 8) gated RMSNorm
    kda_rmsnorm_gate<<<T_LEN * NH, HD, 0, stream>>>(obuf, ogbuf, norm_w);

    // 9) output projection (WMMA) -> d_out [T, HID] f32
    cvt(obuf, o_bf, T_LEN * PROJ);
    gemm(o_bf, wo_bf, out, T_LEN, HID, PROJ);
}
